// LyricSTM_3788161155465
// MI455X (gfx1250) — matrix operand order repaired and checked by reading, not compiled, not run on hardware
//
#include <hip/hip_runtime.h>

typedef __attribute__((ext_vector_type(16))) __bf16 v16bf;
typedef __attribute__((ext_vector_type(8)))  float  v8f;
typedef __attribute__((ext_vector_type(16))) unsigned short v16u;
typedef __attribute__((ext_vector_type(8)))  unsigned short v8u;

#define T_STEPS 512
#define NB   512      // batch
#define NF   128      // x features
#define NG   12       // gen features
#define NH   768      // hidden
#define ND1  512      // head hidden
#define NA   128      // head out
#define K1   (NF + NH)    // 896  (LSTM1 fused K)
#define K2   (2 * NH)     // 1536 (LSTM2 fused K)
#define KZ   800          // H+G padded to mult of 32
#define N4H  (4 * NH)     // 3072

__device__ __forceinline__ unsigned short f2bf(float f) {
  unsigned int u = __float_as_uint(f);
  u += 0x7fffu + ((u >> 16) & 1u);          // round-to-nearest-even
  return (unsigned short)(u >> 16);
}
__device__ __forceinline__ float sigf(float x) { return 1.f / (1.f + __expf(-x)); }

__device__ __forceinline__ v16bf load_afrag(const unsigned short* row, int k, int koffA) {
  union { v16u u; v16bf b; } a;
  v8u alo = *reinterpret_cast<const v8u*>(row + k + koffA);
  v8u ahi = *reinterpret_cast<const v8u*>(row + k + 8 + koffA);
#pragma unroll
  for (int i = 0; i < 8; ++i) { a.u[i] = alo[i]; a.u[8 + i] = ahi[i]; }
  return a.b;
}

// ---------------- setup kernels ----------------

// dst[n][k] (N x Kt, bf16) = k<ki ? Wi[n][k] : (Wh ? Wh[n][k-ki] : 0)
__global__ void build_w_kernel(const float* __restrict__ Wi, int ki,
                               const float* __restrict__ Wh, int kh,
                               unsigned short* __restrict__ dst, int Kt,
                               long long total) {
  long long idx = (long long)blockIdx.x * blockDim.x + threadIdx.x;
  if (idx >= total) return;
  int n = (int)(idx / Kt), k = (int)(idx % Kt);
  float v = 0.f;
  if (k < ki)                       v = Wi[(size_t)n * ki + k];
  else if (Wh && (k - ki) < kh)     v = Wh[(size_t)n * kh + (k - ki)];
  dst[idx] = f2bf(v);
}

__global__ void addvec_kernel(const float* __restrict__ a, const float* __restrict__ b,
                              float* __restrict__ o, int n) {
  int i = blockIdx.x * blockDim.x + threadIdx.x;
  if (i < n) o[i] = a[i] + b[i];
}

__global__ void zero_u16_kernel(unsigned short* p, long long n) {
  long long i = (long long)blockIdx.x * blockDim.x + threadIdx.x;
  if (i < n) p[i] = 0;
}
__global__ void zero_f32_kernel(float* p, long long n) {
  long long i = (long long)blockIdx.x * blockDim.x + threadIdx.x;
  if (i < n) p[i] = 0.f;
}

// ---------------- per-step helpers ----------------

// x[t] (B x 140 f32) -> act1[:,0:128] bf16 and z[:,768:780] bf16
__global__ void pack_x_kernel(const float* __restrict__ x,
                              unsigned short* __restrict__ act1,
                              unsigned short* __restrict__ z, int t) {
  int idx = blockIdx.x * blockDim.x + threadIdx.x;   // < NB*(NF+NG)
  int b = idx / (NF + NG), c = idx % (NF + NG);
  float v = x[(size_t)t * NB * (NF + NG) + idx];
  if (c < NF) act1[(size_t)b * K1 + c]             = f2bf(v);
  else        z   [(size_t)b * KZ + NH + (c - NF)] = f2bf(v);
}

// h1 (act2[:,0:768]) -> act1[:,128:896];  h2 (z[:,0:768]) -> act2[:,768:1536]
__global__ void copy_h_kernel(unsigned short* __restrict__ act1,
                              unsigned short* __restrict__ act2,
                              const unsigned short* __restrict__ z) {
  int idx = blockIdx.x * blockDim.x + threadIdx.x;   // < NB*NH
  int b = idx / NH, j = idx % NH;
  act1[(size_t)b * K1 + NF + j] = act2[(size_t)b * K2 + j];
  act2[(size_t)b * K2 + NH + j] = z[(size_t)b * KZ + j];
}

// ---------------- LSTM gates GEMM + fused cell update ----------------
// gates = act(BxK) * Wc(3072xK)^T + bsum
// Each wave owns a 32(batch) x 16(hidden-col) tile of ALL FOUR gates:
//   2 A-fragments x 4 gate B-fragments = 8 WMMAs per 32-wide K step.
// grid: (NB/32)*(NH/16) = 768 waves = 96 blocks x 256 threads (exact; EXEC all-1s)
__global__ void lstm_gates_kernel(const unsigned short* __restrict__ act,
                                  const unsigned short* __restrict__ Wc,
                                  const float* __restrict__ bsum,
                                  float* __restrict__ cstate,           // NB x NH
                                  unsigned short* __restrict__ hdst, int ldh,
                                  int K) {
  const int lane = threadIdx.x & 31;
  const int wave = blockIdx.x * (blockDim.x >> 5) + (threadIdx.x >> 5);
  const int tn = wave % (NH / 16);
  const int tm = wave / (NH / 16);          // 0..15, 32 batch rows each
  const int half = lane >> 4;
  const int l15  = lane & 15;
  const int koffA = half * 16;
  const int koffB = half * 16;

  const unsigned short* aptr0 = act + (size_t)(tm * 32 + l15) * K;
  const unsigned short* aptr1 = aptr0 + (size_t)16 * K;
  const unsigned short* wptr[4];
#pragma unroll
  for (int g = 0; g < 4; ++g)
    wptr[g] = Wc + (size_t)(g * NH + tn * 16 + l15) * K;

  v8f acc[2][4];
#pragma unroll
  for (int mi = 0; mi < 2; ++mi)
#pragma unroll
    for (int g = 0; g < 4; ++g)
      acc[mi][g] = (v8f){0.f,0.f,0.f,0.f,0.f,0.f,0.f,0.f};

  for (int k = 0; k < K; k += 32) {
    v16bf a0 = load_afrag(aptr0, k, koffA);
    v16bf a1 = load_afrag(aptr1, k, koffA);
    __builtin_prefetch(aptr0 + k + 32, 0, 1);   // global_prefetch_b8 (speculative-safe)
#pragma unroll
    for (int g = 0; g < 4; ++g) {
      union { v16u u; v16bf b; } bw;
      bw.u = *reinterpret_cast<const v16u*>(wptr[g] + k + koffB);
      acc[0][g] = __builtin_amdgcn_wmma_f32_16x16x32_bf16(
          false, a0, false, bw.b, (short)0, acc[0][g], false, false);
      acc[1][g] = __builtin_amdgcn_wmma_f32_16x16x32_bf16(
          false, a1, false, bw.b, (short)0, acc[1][g], false, false);
    }
  }

  const int col = tn * 16 + l15;
  const float bi = bsum[0 * NH + col];
  const float bf = bsum[1 * NH + col];
  const float bg = bsum[2 * NH + col];
  const float bo = bsum[3 * NH + col];
#pragma unroll
  for (int mi = 0; mi < 2; ++mi) {
    const int rbase = tm * 32 + mi * 16 + half * 8;
#pragma unroll
    for (int v = 0; v < 8; ++v) {
      const int rm = rbase + v;
      float gi = sigf (acc[mi][0][v] + bi);
      float gf = sigf (acc[mi][1][v] + bf);
      float gg = tanhf(acc[mi][2][v] + bg);
      float go = sigf (acc[mi][3][v] + bo);
      size_t ci = (size_t)rm * NH + col;
      float cn  = gf * cstate[ci] + gi * gg;
      cstate[ci] = cn;
      hdst[(size_t)rm * ldh + col] = f2bf(go * tanhf(cn));
    }
  }
}

// ---------------- head GEMM (32x16 tile per wave) ----------------
// out = A(NBxK) * W(NxK)^T + bias ; selu_mode: 1 => SELU -> bf16, 0 => f32 out
__global__ void head_gemm_kernel(const unsigned short* __restrict__ A, int lda,
                                 const unsigned short* __restrict__ W,
                                 const float* __restrict__ bias,
                                 int N, int K, int selu_mode,
                                 unsigned short* __restrict__ obf,
                                 float* __restrict__ of32, int ldo) {
  const int lane = threadIdx.x & 31;
  const int wave = blockIdx.x * (blockDim.x >> 5) + (threadIdx.x >> 5);
  const int tiles_n = N / 16;
  const int tm = wave / tiles_n, tn = wave % tiles_n;
  const int half = lane >> 4, l15 = lane & 15;
  const int koffA = half * 16, koffB = half * 16;

  const unsigned short* aptr0 = A + (size_t)(tm * 32 + l15) * lda;
  const unsigned short* aptr1 = aptr0 + (size_t)16 * lda;
  const unsigned short* wptr  = W + (size_t)(tn * 16 + l15) * K;

  v8f acc0 = (v8f){0.f,0.f,0.f,0.f,0.f,0.f,0.f,0.f};
  v8f acc1 = (v8f){0.f,0.f,0.f,0.f,0.f,0.f,0.f,0.f};
  for (int k = 0; k < K; k += 32) {
    v16bf a0 = load_afrag(aptr0, k, koffA);
    v16bf a1 = load_afrag(aptr1, k, koffA);
    union { v16u u; v16bf b; } bw;
    bw.u = *reinterpret_cast<const v16u*>(wptr + k + koffB);
    acc0 = __builtin_amdgcn_wmma_f32_16x16x32_bf16(
        false, a0, false, bw.b, (short)0, acc0, false, false);
    acc1 = __builtin_amdgcn_wmma_f32_16x16x32_bf16(
        false, a1, false, bw.b, (short)0, acc1, false, false);
  }

  const int col = tn * 16 + l15;
  const float b = bias[col];
  const float SC = 1.0507009873554805f, AL = 1.6732632423543772f;
#pragma unroll
  for (int mi = 0; mi < 2; ++mi) {
    const v8f& acc = mi ? acc1 : acc0;
    const int rbase = tm * 32 + mi * 16 + half * 8;
#pragma unroll
    for (int v = 0; v < 8; ++v) {
      const int rm = rbase + v;
      float val = acc[v] + b;
      if (selu_mode) {
        val = (val > 0.f) ? SC * val : SC * AL * (__expf(val) - 1.f);
        obf[(size_t)rm * ldo + col] = f2bf(val);
      } else {
        of32[(size_t)rm * ldo + col] = val;
      }
    }
  }
}

// ---------------- launch ----------------
extern "C" void kernel_launch(void* const* d_in, const int* in_sizes, int n_in,
                              void* d_out, int out_size, void* d_ws, size_t ws_size,
                              hipStream_t stream) {
  const float* x     = (const float*)d_in[0];
  const float* W_ih1 = (const float*)d_in[1];
  const float* W_hh1 = (const float*)d_in[2];
  const float* b_ih1 = (const float*)d_in[3];
  const float* b_hh1 = (const float*)d_in[4];
  const float* W_ih2 = (const float*)d_in[5];
  const float* W_hh2 = (const float*)d_in[6];
  const float* b_ih2 = (const float*)d_in[7];
  const float* b_hh2 = (const float*)d_in[8];
  const float* l1_w  = (const float*)d_in[9];
  const float* l1_b  = (const float*)d_in[10];
  const float* l2_w  = (const float*)d_in[11];
  const float* l2_b  = (const float*)d_in[12];
  float* out = (float*)d_out;

  char* w = (char*)d_ws;
  size_t off = 0;
  auto alloc = [&](size_t bytes) -> void* {
    void* p = w + off;
    off += (bytes + 255) & ~(size_t)255;
    return p;
  };
  unsigned short* WC1 = (unsigned short*)alloc((size_t)N4H * K1 * 2);  // 3072x896
  unsigned short* WC2 = (unsigned short*)alloc((size_t)N4H * K2 * 2);  // 3072x1536
  unsigned short* L1W = (unsigned short*)alloc((size_t)ND1 * KZ * 2);  // 512x800
  unsigned short* L2W = (unsigned short*)alloc((size_t)NA * ND1 * 2);  // 128x512
  float* BS1          = (float*)alloc((size_t)N4H * 4);
  float* BS2          = (float*)alloc((size_t)N4H * 4);
  unsigned short* ACT1= (unsigned short*)alloc((size_t)NB * K1 * 2);   // [x | h1]
  unsigned short* ACT2= (unsigned short*)alloc((size_t)NB * K2 * 2);   // [h1 | h2]
  unsigned short* Z   = (unsigned short*)alloc((size_t)NB * KZ * 2);   // [h2 | gen | pad]
  unsigned short* Y1  = (unsigned short*)alloc((size_t)NB * ND1 * 2);
  float* C1           = (float*)alloc((size_t)NB * NH * 4);
  float* C2           = (float*)alloc((size_t)NB * NH * 4);
  (void)in_sizes; (void)n_in; (void)out_size; (void)ws_size;

  const dim3 blk(256);
  auto nb = [](long long n) { return dim3((unsigned)((n + 255) / 256)); };

  // --- init (re-done every launch: deterministic, graph-capture safe) ---
  build_w_kernel<<<nb((long long)N4H * K1), blk, 0, stream>>>(W_ih1, NF, W_hh1, NH, WC1, K1, (long long)N4H * K1);
  build_w_kernel<<<nb((long long)N4H * K2), blk, 0, stream>>>(W_ih2, NH, W_hh2, NH, WC2, K2, (long long)N4H * K2);
  build_w_kernel<<<nb((long long)ND1 * KZ), blk, 0, stream>>>(l1_w, NH + NG, nullptr, 0, L1W, KZ, (long long)ND1 * KZ);
  build_w_kernel<<<nb((long long)NA * ND1), blk, 0, stream>>>(l2_w, ND1, nullptr, 0, L2W, ND1, (long long)NA * ND1);
  addvec_kernel<<<nb(N4H), blk, 0, stream>>>(b_ih1, b_hh1, BS1, N4H);
  addvec_kernel<<<nb(N4H), blk, 0, stream>>>(b_ih2, b_hh2, BS2, N4H);
  zero_u16_kernel<<<nb((long long)NB * K1), blk, 0, stream>>>(ACT1, (long long)NB * K1);
  zero_u16_kernel<<<nb((long long)NB * K2), blk, 0, stream>>>(ACT2, (long long)NB * K2);
  zero_u16_kernel<<<nb((long long)NB * KZ), blk, 0, stream>>>(Z,    (long long)NB * KZ);
  zero_f32_kernel<<<nb((long long)NB * NH), blk, 0, stream>>>(C1,   (long long)NB * NH);
  zero_f32_kernel<<<nb((long long)NB * NH), blk, 0, stream>>>(C2,   (long long)NB * NH);

  const dim3 gatesGrid((NB / 32) * (NH / 16) / 8);    // 768 waves -> 96 blocks
  const dim3 l1Grid((NB / 32) * (ND1 / 16) / 8);      // 512 waves -> 64 blocks
  const dim3 l2Grid((NB / 32) * (NA / 16) / 8);       // 128 waves -> 16 blocks

  for (int t = 0; t < T_STEPS; ++t) {
    pack_x_kernel<<<nb((long long)NB * (NF + NG)), blk, 0, stream>>>(x, ACT1, Z, t);
    // LSTM1: reads ACT1=[x_t|h1_{t-1}], writes h1_t -> ACT2[:,0:768], updates C1
    lstm_gates_kernel<<<gatesGrid, blk, 0, stream>>>(ACT1, WC1, BS1, C1, ACT2, K2, K1);
    // LSTM2: reads ACT2=[h1_t|h2_{t-1}], writes h2_t -> Z[:,0:768], updates C2
    lstm_gates_kernel<<<gatesGrid, blk, 0, stream>>>(ACT2, WC2, BS2, C2, Z, KZ, K2);
    // fan h out for next step (separate kernel: no RAW hazard inside GEMMs)
    copy_h_kernel<<<nb((long long)NB * NH), blk, 0, stream>>>(ACT1, ACT2, Z);
    // head: selu(z @ l1_w^T + l1_b) -> Y1 (bf16)
    head_gemm_kernel<<<l1Grid, blk, 0, stream>>>(Z, KZ, L1W, l1_b, ND1, KZ, 1, Y1, nullptr, ND1);
    // head: Y1 @ l2_w^T + l2_b -> out[t] (f32)
    head_gemm_kernel<<<l2Grid, blk, 0, stream>>>(Y1, ND1, L2W, l2_b, NA, ND1, 0, nullptr,
                                                 out + (size_t)t * NB * NA, NA);
  }
}
